// AttentionConv2d_82755429859715
// MI455X (gfx1250) — matrix operand order repaired and checked by reading, not compiled, not run on hardware
//
#include <hip/hip_runtime.h>
#include <hip/hip_bf16.h>

#define DEVINL __device__ __forceinline__

typedef __attribute__((ext_vector_type(8)))  float          v8f;
typedef __attribute__((ext_vector_type(4)))  float          v4f;
typedef __attribute__((ext_vector_type(16))) __bf16         v16bf;
typedef unsigned short u16t;
typedef __attribute__((ext_vector_type(16))) unsigned short us16;
typedef __attribute__((ext_vector_type(8)))  unsigned short us8;
typedef __attribute__((ext_vector_type(4)))  unsigned int   u32x4;
typedef __attribute__((ext_vector_type(8)))  int            i32x8;
typedef __attribute__((ext_vector_type(4)))  int            i32x4;

// CDNA5 data-movement feature gates (compile-safe: existence probed)
#if __has_builtin(__builtin_amdgcn_tensor_load_to_lds) && __has_builtin(__builtin_amdgcn_s_wait_tensorcnt)
#define USE_TDM 1
#else
#define USE_TDM 0
#endif
#if !USE_TDM && __has_builtin(__builtin_amdgcn_global_load_async_to_lds_b128) && __has_builtin(__builtin_amdgcn_s_wait_asynccnt)
#define USE_ASYNC 1
#else
#define USE_ASYNC 0
#endif

// Problem sizes: B=32, CIN=256, N=H*W=1024, DK=DV=128, HEADS=8, DKH=DVH=16, OUT=256
static constexpr float QSCALE = 0.25f * 1.4426950408889634f; // DKH^-0.5 * log2(e): softmax uses exp2

DEVINL u16t f2bf(float f) {                     // fp32 -> bf16, round-to-nearest-even
  unsigned u = __builtin_bit_cast(unsigned, f);
  u = (u + 0x7FFFu + ((u >> 16) & 1u)) >> 16;
  return (u16t)u;
}

DEVINL us16 cat8(us8 lo, us8 hi) {
  return __builtin_shufflevector(lo, hi, 0,1,2,3,4,5,6,7,8,9,10,11,12,13,14,15);
}

DEVINL v8f wmma_bf16(us16 a, us16 b, v8f c) {
  return __builtin_amdgcn_wmma_f32_16x16x32_bf16(
      false, __builtin_bit_cast(v16bf, a),
      false, __builtin_bit_cast(v16bf, b),
      (short)0, c, false, false);
}

// ---------------------------------------------------------------------------
// Kernel 1: fused projection GEMM.
//   Out[384+128, 1024] = [w_qkv ; w_out] (fp32) @ x[b] (fp32), per batch.
//   mtile 0 -> q (bf16, scaled, layout [b][n][128c])  ("A-ready" for stage 2)
//   mtile 1 -> k (bf16, layout [b][128c][n])
//   mtile 2 -> v (bf16, layout [b][128c][n])
//   mtile 3 -> conv_out (fp32 + bias) into d_out channels [0,128)
// Workgroup: 256 thr (8 waves), tile 128M x 128N, wave tile 64x32, K-step 32.
// ---------------------------------------------------------------------------
__global__ __launch_bounds__(256)
void k_proj(const float* __restrict__ x,
            const float* __restrict__ w_qkv, const float* __restrict__ b_qkv,
            const float* __restrict__ w_out, const float* __restrict__ b_out,
            float* __restrict__ out,
            u16t* __restrict__ qb, u16t* __restrict__ kb, u16t* __restrict__ vb)
{
  __shared__ __attribute__((aligned(32))) u16t Wt[128 * 32]; // [m][k] bf16
  __shared__ __attribute__((aligned(32))) u16t Xt[128 * 32]; // [n][k] bf16 (transposed)

  const int t     = threadIdx.x;
  const int ntile = blockIdx.x;   // 0..7
  const int mtile = blockIdx.y;   // 0=q 1=k 2=v 3=conv
  const int bz    = blockIdx.z;   // batch
  const int n0    = ntile * 128;

  const float* wsrc = (mtile < 3) ? (w_qkv + (size_t)mtile * 128 * 256) : w_out;

  const int lane = t & 31, wave = t >> 5;
  const int wm = wave >> 2, wn = wave & 3;      // 2 x 4 wave grid
  const int half = lane >> 4, ln = lane & 15;

  const v8f zv = {0.f,0.f,0.f,0.f,0.f,0.f,0.f,0.f};
  v8f acc[4][2];
#pragma unroll
  for (int i = 0; i < 4; ++i)
#pragma unroll
    for (int j = 0; j < 2; ++j) acc[i][j] = zv;

  for (int ks = 0; ks < 8; ++ks) {
    // ---- stage W tile [128m][32k] (fp32 -> bf16)
    {
      const int r = t >> 1, kc = (t & 1) * 16;
      const float* wrow = wsrc + (size_t)r * 256 + ks * 32 + kc;
      v4f a0 = *(const v4f*)(wrow + 0);
      v4f a1 = *(const v4f*)(wrow + 4);
      v4f a2 = *(const v4f*)(wrow + 8);
      v4f a3 = *(const v4f*)(wrow + 12);
      us8 p0, p1;
#pragma unroll
      for (int e = 0; e < 4; ++e) {
        p0[e]     = f2bf(a0[e]);
        p0[4 + e] = f2bf(a1[e]);
        p1[e]     = f2bf(a2[e]);
        p1[4 + e] = f2bf(a3[e]);
      }
      *(us8*)&Wt[r * 32 + kc]     = p0;
      *(us8*)&Wt[r * 32 + kc + 8] = p1;
    }
    // ---- stage X tile transposed -> [128n][32k] (fp32 -> bf16)
    {
      const int cl = t & 31, ng = t >> 5;
      const float* xrow = x + ((size_t)bz * 256 + ks * 32 + cl) * 1024 + n0 + ng * 16;
      if (ks < 7) __builtin_prefetch(xrow + 32 * 1024, 0, 0);
      v4f a0 = *(const v4f*)(xrow + 0);
      v4f a1 = *(const v4f*)(xrow + 4);
      v4f a2 = *(const v4f*)(xrow + 8);
      v4f a3 = *(const v4f*)(xrow + 12);
#pragma unroll
      for (int e = 0; e < 4; ++e) {
        Xt[(ng * 16 + e     ) * 32 + cl] = f2bf(a0[e]);
        Xt[(ng * 16 + 4 + e ) * 32 + cl] = f2bf(a1[e]);
        Xt[(ng * 16 + 8 + e ) * 32 + cl] = f2bf(a2[e]);
        Xt[(ng * 16 + 12 + e) * 32 + cl] = f2bf(a3[e]);
      }
    }
    __syncthreads();

    // ---- fragments + 8 WMMAs
    const int k0l = 8 * half;
    us16 afr[4], bfr[2];
#pragma unroll
    for (int i = 0; i < 4; ++i) {
      const int row = wm * 64 + i * 16 + ln;
      us8 lo = *(const us8*)&Wt[row * 32 + k0l];
      us8 hi = *(const us8*)&Wt[row * 32 + k0l + 16];
      afr[i] = cat8(lo, hi);
    }
#pragma unroll
    for (int j = 0; j < 2; ++j) {
      const int col = wn * 32 + j * 16 + ln;
      bfr[j] = cat8(*(const us8*)&Xt[col * 32 + 8 * half], *(const us8*)&Xt[col * 32 + 8 * half + 16]);
    }
#pragma unroll
    for (int i = 0; i < 4; ++i)
#pragma unroll
      for (int j = 0; j < 2; ++j)
        acc[i][j] = wmma_bf16(afr[i], bfr[j], acc[i][j]);
    __syncthreads();
  }

  // ---- epilogue: bias + route
  u16t* dstb = (mtile == 0) ? qb : (mtile == 1) ? kb : vb;
#pragma unroll
  for (int i = 0; i < 4; ++i) {
#pragma unroll
    for (int j = 0; j < 2; ++j) {
      const int colg = n0 + wn * 32 + j * 16 + ln;
#pragma unroll
      for (int jj = 0; jj < 8; ++jj) {
        const int rowl = wm * 64 + i * 16 + jj + 8 * half;   // 0..127
        float v = acc[i][j][jj];
        if (mtile == 3) {
          v += b_out[rowl];
          out[((size_t)bz * 256 + rowl) * 1024 + colg] = v;  // channels [0,128)
        } else if (mtile == 0) {
          v = (v + b_qkv[rowl]) * QSCALE;
          qb[((size_t)bz * 1024 + colg) * 128 + rowl] = f2bf(v);   // [b][n][c]
        } else {
          v += b_qkv[mtile * 128 + rowl];
          dstb[((size_t)bz * 128 + rowl) * 1024 + colg] = f2bf(v); // [b][c][n]
        }
      }
    }
  }
}

// ---------------------------------------------------------------------------
// Kernel 2: attention per (batch, head).  Flash-style online softmax.
//   K,V for the head (16 x 1024 bf16 each) resident in LDS.
//   V staging goes through the Tensor Data Mover when available (32 KB
//   contiguous, wave-uniform: ideal TDM shape), else async-to-LDS, else VALU.
// ---------------------------------------------------------------------------
__global__ __launch_bounds__(256)
void k_attn(const u16t* __restrict__ qb, const u16t* __restrict__ kbuf,
            const u16t* __restrict__ vbuf, u16t* __restrict__ ab)
{
  __shared__ __attribute__((aligned(32))) u16t Ks[1024 * 16]; // [key][d]
  __shared__ __attribute__((aligned(32))) u16t Vs[16 * 1024]; // [d][key]
  __shared__ __attribute__((aligned(32))) u16t Ps[8][16 * 32]; // per-wave P staging

  const int t  = threadIdx.x;
  const int h  = blockIdx.x;    // head
  const int bz = blockIdx.y;    // batch
  const size_t headoff = ((size_t)bz * 128 + h * 16) * 1024;

  // ---- stage V ([d][key] natural layout, 32 KB contiguous)
#if USE_TDM
  if (t < 32) {
    // D# for a 1-D 16384-element (2B) tile: global -> LDS.
    const unsigned long long ga = (unsigned long long)(vbuf + headoff);
    const unsigned lds_off = (unsigned)(size_t)&Vs[0];   // LDS aperture: addr[31:0] = LDS offset
    u32x4 g0;
    g0[0] = 1u;                                   // count=1, user descriptor
    g0[1] = lds_off;                              // lds_addr (bytes)
    g0[2] = (unsigned)(ga & 0xFFFFFFFFull);       // global_addr[31:0]
    g0[3] = (unsigned)((ga >> 32) & 0x1FFFFFFull) // global_addr[56:32]
            | 0x80000000u;                        // type=2 ("image")
    i32x8 g1;
    g1[0] = (int)(1u << 16);                      // workgroup_mask=0, data_size=1 (2B)
    g1[1] = (int)(16384u << 16);                  // tensor_dim0[15:0] << 16
    g1[2] = (int)(1u << 16);                      // tensor_dim0[31:16]=0 | tensor_dim1=1
    g1[3] = (int)(16384u << 16);                  // tile_dim0 = 16384
    g1[4] = 0;                                    // tile_dim1=0 (unused), tile_dim2=0
    g1[5] = 16384;                                // tensor_dim0_stride[31:0]
    g1[6] = 0;
    g1[7] = 0;
    const i32x4 gz  = {0, 0, 0, 0};
    const i32x8 gz8 = {0, 0, 0, 0, 0, 0, 0, 0};
    __builtin_amdgcn_tensor_load_to_lds(g0, g1, gz, gz, gz8, 0); // 6-arg toolchain form
    __builtin_amdgcn_s_wait_tensorcnt(0);
  }
#elif USE_ASYNC
  {
    u16t* vsrc = (u16t*)vbuf + headoff + (size_t)t * 64;
#pragma unroll
    for (int i = 0; i < 64; i += 8)
      __builtin_amdgcn_global_load_async_to_lds_b128(
          (__attribute__((address_space(1))) void*)(vsrc + i),
          (__attribute__((address_space(3))) void*)&Vs[t * 64 + i], 0, 0);
    __builtin_amdgcn_s_wait_asynccnt(0);
  }
#else
  {
    const u16t* vsrc = vbuf + headoff + (size_t)t * 64;
#pragma unroll
    for (int i = 0; i < 64; i += 8)
      *(us8*)&Vs[t * 64 + i] = *(const us8*)(vsrc + i);
  }
#endif

  // ---- stage K transposed -> [key][d]
  {
    const int d = t & 15, g = t >> 4;  // g = 0..15, each covers 64 keys
    const u16t* ksrc = kbuf + headoff + (size_t)d * 1024 + g * 64;
#pragma unroll
    for (int i = 0; i < 64; i += 8) {
      us8 kv = *(const us8*)(ksrc + i);
#pragma unroll
      for (int e = 0; e < 8; ++e) Ks[(g * 64 + i + e) * 16 + d] = kv[e];
    }
  }
  __syncthreads();

  const int lane = t & 31, wave = t >> 5;
  const int half = lane >> 4, ln = lane & 15;
  const us8  z8  = {0,0,0,0,0,0,0,0};
  const us16 z16 = cat8(z8, z8);
  const v8f  zv  = {0.f,0.f,0.f,0.f,0.f,0.f,0.f,0.f};

  for (int qt = wave; qt < 64; qt += 8) {
    const int q0 = qt * 16;
    // Q A-fragment straight from global (q stored [b][n][c], 16B per lane),
    // K-halves 16..31 are zero padding (head dim is 16).
    us8 qlo = *(const us8*)&qb[((size_t)bz * 1024 + q0 + ln) * 128 + h * 16 + 8 * half];
    const us16 qfr = cat8(qlo, z8);

    v8f o = zv;
    float m[8], l[8];
#pragma unroll
    for (int j = 0; j < 8; ++j) { m[j] = -1e30f; l[j] = 0.f; }

    for (int kbk = 0; kbk < 32; ++kbk) {
      const int key0 = kbk * 32;
      // B-fragments for S: elements 8..15 carry the zero-padded K range
      us16 kf0 = cat8(*(const us8*)&Ks[(key0 + ln) * 16 + 8 * half], z8);
      us16 kf1 = cat8(*(const us8*)&Ks[(key0 + 16 + ln) * 16 + 8 * half], z8);
      v8f s0 = wmma_bf16(qfr, kf0, zv);   // S[q][key0..key0+15]
      v8f s1 = wmma_bf16(qfr, kf1, zv);   // S[q][key0+16..key0+31]

      // online softmax update (exp2 domain; log2e folded into q scale)
#pragma unroll
      for (int j = 0; j < 8; ++j) {
        float tmax = fmaxf(s0[j], s1[j]);
#pragma unroll
        for (int d = 1; d < 16; d <<= 1)
          tmax = fmaxf(tmax, __shfl_xor(tmax, d, 16));
        const float mn   = fmaxf(m[j], tmax);
        const float corr = exp2f(m[j] - mn);
        const float p0   = exp2f(s0[j] - mn);
        const float p1   = exp2f(s1[j] - mn);
        l[j] = l[j] * corr + p0 + p1;     // per-lane partial row sum
        o[j] = o[j] * corr;
        m[j] = mn;
        const int prow = j + 8 * half;    // C-layout row
        Ps[wave][prow * 32 + ln]      = f2bf(p0);
        Ps[wave][prow * 32 + 16 + ln] = f2bf(p1);
      }
      // P A-fragment (wave-private LDS restage; in-wave DS ops are in order)
      us8 plo = *(const us8*)&Ps[wave][ln * 32 + 8 * half];
      us8 phi = *(const us8*)&Ps[wave][ln * 32 + 8 * half + 16];
      const us16 pfr = cat8(plo, phi);
      // V B-fragment: full K=32 utilization
      const us16 vfr = cat8(*(const us8*)&Vs[ln * 1024 + key0 + 8 * half], *(const us8*)&Vs[ln * 1024 + key0 + 8 * half + 16]);
      o = wmma_bf16(pfr, vfr, o);         // O[q][d] accumulate
    }

    // finalize: row-sum reduce, normalize, pack 8 consecutive n per lane
    us8 ov = z8;
#pragma unroll
    for (int j = 0; j < 8; ++j) {
      float ls = l[j];
#pragma unroll
      for (int d = 1; d < 16; d <<= 1) ls += __shfl_xor(ls, d, 16);
      ov[j] = f2bf(o[j] / ls);
    }
    *(us8*)&ab[headoff + (size_t)ln * 1024 + q0 + 8 * half] = ov; // [b][c][n]
  }
}

// ---------------------------------------------------------------------------
// Kernel 3: attention output projection.
//   out[ch 128..255] = w_attn[128,128] @ attn[128,1024] + b_attn, per batch.
// ---------------------------------------------------------------------------
__global__ __launch_bounds__(256)
void k_attnproj(const u16t* __restrict__ ab,
                const float* __restrict__ w_attn, const float* __restrict__ b_attn,
                float* __restrict__ out)
{
  __shared__ __attribute__((aligned(32))) u16t Wt[128 * 32];
  __shared__ __attribute__((aligned(32))) u16t Xt[128 * 32];

  const int t     = threadIdx.x;
  const int ntile = blockIdx.x;   // 0..7
  const int bz    = blockIdx.y;
  const int n0    = ntile * 128;

  const int lane = t & 31, wave = t >> 5;
  const int wm = wave >> 2, wn = wave & 3;
  const int half = lane >> 4, ln = lane & 15;

  const v8f zv = {0.f,0.f,0.f,0.f,0.f,0.f,0.f,0.f};
  v8f acc[4][2];
#pragma unroll
  for (int i = 0; i < 4; ++i)
#pragma unroll
    for (int j = 0; j < 2; ++j) acc[i][j] = zv;

  for (int ks = 0; ks < 4; ++ks) {
    { // W tile (fp32 -> bf16)
      const int r = t >> 1, kc = (t & 1) * 16;
      const float* wrow = w_attn + (size_t)r * 128 + ks * 32 + kc;
      v4f a0 = *(const v4f*)(wrow + 0);
      v4f a1 = *(const v4f*)(wrow + 4);
      v4f a2 = *(const v4f*)(wrow + 8);
      v4f a3 = *(const v4f*)(wrow + 12);
      us8 p0, p1;
#pragma unroll
      for (int e = 0; e < 4; ++e) {
        p0[e]     = f2bf(a0[e]);
        p0[4 + e] = f2bf(a1[e]);
        p1[e]     = f2bf(a2[e]);
        p1[4 + e] = f2bf(a3[e]);
      }
      *(us8*)&Wt[r * 32 + kc]     = p0;
      *(us8*)&Wt[r * 32 + kc + 8] = p1;
    }
    { // attn tile transposed (already bf16)
      const int cl = t & 31, ng = t >> 5;
      const u16t* arow = ab + ((size_t)bz * 128 + ks * 32 + cl) * 1024 + n0 + ng * 16;
      us8 v0 = *(const us8*)(arow + 0);
      us8 v1 = *(const us8*)(arow + 8);
#pragma unroll
      for (int e = 0; e < 8; ++e) {
        Xt[(ng * 16 + e    ) * 32 + cl] = v0[e];
        Xt[(ng * 16 + 8 + e) * 32 + cl] = v1[e];
      }
    }
    __syncthreads();

    const int k0l = 8 * half;
    us16 afr[4], bfr[2];
#pragma unroll
    for (int i = 0; i < 4; ++i) {
      const int row = wm * 64 + i * 16 + ln;
      us8 lo = *(const us8*)&Wt[row * 32 + k0l];
      us8 hi = *(const us8*)&Wt[row * 32 + k0l + 16];
      afr[i] = cat8(lo, hi);
    }
#pragma unroll
    for (int j = 0; j < 2; ++j) {
      const int col = wn * 32 + j * 16 + ln;
      bfr[j] = cat8(*(const us8*)&Xt[col * 32 + 8 * half], *(const us8*)&Xt[col * 32 + 8 * half + 16]);
    }
#pragma unroll
    for (int i = 0; i < 4; ++i)
#pragma unroll
      for (int j = 0; j < 2; ++j)
        acc[i][j] = wmma_bf16(afr[i], bfr[j], acc[i][j]);
    __syncthreads();
  }

#pragma unroll
  for (int i = 0; i < 4; ++i)
#pragma unroll
    for (int j = 0; j < 2; ++j) {
      const int colg = n0 + wn * 32 + j * 16 + ln;
#pragma unroll
      for (int jj = 0; jj < 8; ++jj) {
        const int rowl = wm * 64 + i * 16 + jj + 8 * half;
        out[((size_t)bz * 256 + 128 + rowl) * 1024 + colg] =
            acc[i][j][jj] + b_attn[rowl];
      }
    }
}

// ---------------------------------------------------------------------------
extern "C" void kernel_launch(void* const* d_in, const int* in_sizes, int n_in,
                              void* d_out, int out_size, void* d_ws, size_t ws_size,
                              hipStream_t stream) {
  const float* x      = (const float*)d_in[0];
  const float* w_qkv  = (const float*)d_in[1];
  const float* b_qkv  = (const float*)d_in[2];
  const float* w_attn = (const float*)d_in[3];
  const float* b_attn = (const float*)d_in[4];
  const float* w_out  = (const float*)d_in[5];
  const float* b_out  = (const float*)d_in[6];
  float* out = (float*)d_out;

  // Workspace: 4 bf16 buffers of 32*128*1024 elements (8 MB each, 32 MB total)
  u16t* ws = (u16t*)d_ws;
  const size_t SEG = (size_t)32 * 128 * 1024;
  u16t* qb = ws;            // [b][n][c]  (A-fragment-ready for attention)
  u16t* kb = ws + SEG;      // [b][c][n]
  u16t* vb = ws + 2 * SEG;  // [b][c][n]
  u16t* ab = ws + 3 * SEG;  // [b][c][n]  attention result pre-projection

  k_proj    <<<dim3(8, 4, 32), 256, 0, stream>>>(x, w_qkv, b_qkv, w_out, b_out, out, qb, kb, vb);
  k_attn    <<<dim3(8, 32),    256, 0, stream>>>(qb, kb, vb, ab);
  k_attnproj<<<dim3(8, 32),    256, 0, stream>>>(ab, w_attn, b_attn, out);
}
